// LoRALinear_8426725835305
// MI455X (gfx1250) — matrix operand order repaired and checked by reading, not compiled, not run on hardware
//
#include <hip/hip_runtime.h>

// ---------------------------------------------------------------------------
// LoRA linear for MI455X (gfx1250, wave32, WMMA):
//   out = x @ W^T + b + 2.0 * ((bf16(x) @ A^T) @ B^T)
//
// Base GEMM (275 GFLOP) as split-bf16 "bf16x3" on v_wmma_f32_16x16x32_bf16
// (x = x_hi + x_lo, W = W_hi + W_lo; hi*hi + lo*hi + hi*lo, fp32 accum):
// ~fp32 fidelity at ~3x the throughput of the native f32 WMMA (K=4 vs K=32).
//
// Block tile 128(M) x 256(N) x 32(K); 8 waves, each owning a 64x64 C tile
// (16 wmma tiles, 48 WMMAs per k-step vs 32 ds_load_b128 -> WMMA-limited,
// not LDS-limited). LDS: 4 bf16 planes at stride 40 (80B, conflict-free
// 16B fragment reads) = 60KB. Pipeline: ds_store(k) | barrier | global
// register-prefetch(k+1) + global_prefetch_b8(k+4) | 48 WMMAs. Grid sweeps
// N fastest so W (67MB) stays resident in the 192MB L2 across M-stripes.
// ---------------------------------------------------------------------------

typedef __attribute__((ext_vector_type(16))) __bf16 v16bf;
typedef __attribute__((ext_vector_type(8)))  __bf16 v8bf;
typedef __attribute__((ext_vector_type(8)))  float  v8f;
typedef __attribute__((ext_vector_type(4)))  float  v4f;

#define T_DIM 8192
#define D_DIM 4096
#define R_DIM 16
#define BM 128
#define BN 256
#define BK 32
#define LSTR 40                  // padded LDS row stride (bf16) -> 80B, conflict-free
#define PLANE_X (BM * LSTR)      // 5120 elems
#define PLANE_W (BN * LSTR)      // 10240 elems
#define NKSTEP (D_DIM / BK)      // 128
#define PFDIST 4                 // global_prefetch distance in k-steps

__device__ __forceinline__ v16bf cat8(v8bf a, v8bf b) {
  return __builtin_shufflevector(a, b, 0,1,2,3,4,5,6,7,8,9,10,11,12,13,14,15);
}

__device__ __forceinline__ v8f wmma_bf16(v16bf a, v16bf b, v8f c) {
  return __builtin_amdgcn_wmma_f32_16x16x32_bf16(
      /*neg_a=*/false, a, /*neg_b=*/false, b, /*c_mod=*/(short)0, c,
      /*reuse_a=*/false, /*reuse_b=*/false);
}

// A fragment (16x32 bf16, MxK), ISA 7.12.2: lane<16 -> row M=lane, K {0..7,16..23};
// lane>=16 -> row M=lane-16, K {8..15,24..31}. Two ds_load_b128.
__device__ __forceinline__ v16bf frag_a_lds(const __bf16* plane, int row0, int lane) {
  const int half = lane >> 4, r = lane & 15;
  const __bf16* base = plane + (row0 + r) * LSTR;
  v8bf p0 = *(const v8bf*)(base + 8 * half);
  v8bf p1 = *(const v8bf*)(base + 16 + 8 * half);
  return cat8(p0, p1);
}

// B fragment (32x16 bf16, KxN): lane<16 -> col N=lane, K {0..7,16..23};
// lane>=16 -> col N=lane-16, K {8..15,24..31}. Two ds_load_b128.
__device__ __forceinline__ v16bf frag_b_lds(const __bf16* plane, int col0, int lane) {
  const int half = lane >> 4, c = lane & 15;
  const __bf16* base = plane + (col0 + c) * LSTR + 8 * half;
  v8bf p0 = *(const v8bf*)(base);
  v8bf p1 = *(const v8bf*)(base + 16);
  return cat8(p0, p1);
}

// ------------------------- global staging (kernel 2) -----------------------
// Each thread owns 8-float row chunks: X tile 128x32 -> 2 chunks/thread,
// W tile 256x32 -> 4 chunks/thread. 32B global loads, 16B ds_store_b128/plane.
struct Stage { v4f xv[2][2]; v4f wv[4][2]; };

__device__ __forceinline__ void stage_load(Stage& s, const float* __restrict__ x,
                                           const float* __restrict__ W,
                                           int m0, int n0, int k0, int tid) {
#pragma unroll
  for (int i = 0; i < 2; ++i) {
    const int f8 = tid + (i << 8);
    const int row = f8 >> 2;
    const int c = (f8 & 3) << 3;
    const float* p = x + (size_t)(m0 + row) * D_DIM + k0 + c;
    s.xv[i][0] = *(const v4f*)(p);
    s.xv[i][1] = *(const v4f*)(p + 4);
  }
#pragma unroll
  for (int i = 0; i < 4; ++i) {
    const int f8 = tid + (i << 8);
    const int row = f8 >> 2;
    const int c = (f8 & 3) << 3;
    const float* p = W + (size_t)(n0 + row) * D_DIM + k0 + c;
    s.wv[i][0] = *(const v4f*)(p);
    s.wv[i][1] = *(const v4f*)(p + 4);
  }
}

__device__ __forceinline__ void split8(v4f a, v4f b, v8bf& h, v8bf& l) {
#pragma unroll
  for (int j = 0; j < 4; ++j) {
    __bf16 ha = (__bf16)a[j];              // RNE high part
    __bf16 hb = (__bf16)b[j];
    h[j]     = ha;
    h[4 + j] = hb;
    l[j]     = (__bf16)(a[j] - (float)ha); // residual low part
    l[4 + j] = (__bf16)(b[j] - (float)hb);
  }
}

__device__ __forceinline__ void stage_store(const Stage& s,
                                            __bf16* Xhi, __bf16* Xlo,
                                            __bf16* Whi, __bf16* Wlo, int tid) {
#pragma unroll
  for (int i = 0; i < 2; ++i) {
    const int f8 = tid + (i << 8);
    const int off = (f8 >> 2) * LSTR + ((f8 & 3) << 3);  // 16B aligned
    v8bf h, l;
    split8(s.xv[i][0], s.xv[i][1], h, l);
    *(v8bf*)(Xhi + off) = h;
    *(v8bf*)(Xlo + off) = l;
  }
#pragma unroll
  for (int i = 0; i < 4; ++i) {
    const int f8 = tid + (i << 8);
    const int off = (f8 >> 2) * LSTR + ((f8 & 3) << 3);
    v8bf h, l;
    split8(s.wv[i][0], s.wv[i][1], h, l);
    *(v8bf*)(Whi + off) = h;
    *(v8bf*)(Wlo + off) = l;
  }
}

// ---------------------------------------------------------------------------
// Kernel 1: T[8192,16] = bf16(x) @ lora_a^T   (bf16 WMMA, fp32 accum)
// ---------------------------------------------------------------------------
__global__ __launch_bounds__(256) void lora_xa_kernel(
    const float* __restrict__ x, const unsigned short* __restrict__ lora_a_u,
    unsigned short* __restrict__ tmat_u) {
  const __bf16* A = (const __bf16*)lora_a_u;
  __bf16* tmat = (__bf16*)tmat_u;
  const int tid = threadIdx.x, lane = tid & 31, wave = tid >> 5;
  const int half = lane >> 4;
  const int rowA = blockIdx.x * 128 + wave * 16 + (lane & 15);
  const int colB = lane & 15;

  v8f acc = {};
  for (int k0 = 0; k0 < D_DIM; k0 += BK) {
    const float* xr = x + (size_t)rowA * D_DIM + k0 + 8 * half;
    v4f x0 = *(const v4f*)(xr);
    v4f x1 = *(const v4f*)(xr + 4);
    v4f x2 = *(const v4f*)(xr + 16);
    v4f x3 = *(const v4f*)(xr + 20);
    v16bf af;
#pragma unroll
    for (int j = 0; j < 4; ++j) {
      af[j]      = (__bf16)x0[j];
      af[4 + j]  = (__bf16)x1[j];
      af[8 + j]  = (__bf16)x2[j];
      af[12 + j] = (__bf16)x3[j];
    }
    const __bf16* ar = A + (size_t)colB * D_DIM + k0 + 8 * half;
    v8bf b0 = *(const v8bf*)ar;
    v8bf b1 = *(const v8bf*)(ar + 16);
    acc = wmma_bf16(af, cat8(b0, b1), acc);
  }
  const int rbase = blockIdx.x * 128 + wave * 16 + 8 * half;
#pragma unroll
  for (int i = 0; i < 8; ++i)
    tmat[(size_t)(rbase + i) * R_DIM + colB] = (__bf16)acc[i];
}

// ---------------------------------------------------------------------------
// Kernel 2: 128x256 tiled bf16x3 GEMM + fused bias + LoRA epilogue.
// 8 waves in 2(M) x 4(N) grid -> 64x64 per wave (4x4 wmma tiles).
// ---------------------------------------------------------------------------
__global__ __launch_bounds__(256) void lora_gemm_kernel(
    const float* __restrict__ x, const float* __restrict__ W,
    const float* __restrict__ bias, const unsigned short* __restrict__ lora_b_u,
    const unsigned short* __restrict__ tmat_u, float* __restrict__ out) {
  __shared__ __bf16 smem[2 * PLANE_X + 2 * PLANE_W];   // 60 KB
  __bf16* Xhi = smem;
  __bf16* Xlo = smem + PLANE_X;
  __bf16* Whi = smem + 2 * PLANE_X;
  __bf16* Wlo = smem + 2 * PLANE_X + PLANE_W;
  const __bf16* lora_b = (const __bf16*)lora_b_u;
  const __bf16* tmat = (const __bf16*)tmat_u;

  const int tid = threadIdx.x;
  const int lane = tid & 31, wave = tid >> 5;
  const int wm = wave & 1, wn = wave >> 1;        // 2 x 4 wave grid
  const int half = lane >> 4;
  const int m0 = blockIdx.y * BM;                 // rows of x / out
  const int n0 = blockIdx.x * BN;                 // rows of W == cols of out

  v8f zero = {};
  v8f acc[4][4];
#pragma unroll
  for (int mi = 0; mi < 4; ++mi)
#pragma unroll
    for (int ni = 0; ni < 4; ++ni) acc[mi][ni] = zero;

  Stage st;
  stage_load(st, x, W, m0, n0, 0, tid);

#pragma unroll 1
  for (int kb = 0; kb < NKSTEP; ++kb) {
    __syncthreads();                               // prev compute done reading LDS
    stage_store(st, Xhi, Xlo, Whi, Wlo, tid);      // f32 -> hi/lo bf16 planes
    __syncthreads();                               // tiles visible
    if (kb + 1 < NKSTEP)                           // register prefetch behind WMMAs
      stage_load(st, x, W, m0, n0, (kb + 1) * BK, tid);
    if (kb + PFDIST < NKSTEP) {                    // deep prefetch into WGP$/L2
      const int kpf = (kb + PFDIST) * BK;
#pragma unroll
      for (int i = 0; i < 2; ++i) {
        const int f8 = tid + (i << 8);
        __builtin_prefetch(x + (size_t)(m0 + (f8 >> 2)) * D_DIM + kpf + ((f8 & 3) << 3), 0, 3);
      }
#pragma unroll
      for (int i = 0; i < 4; ++i) {
        const int f8 = tid + (i << 8);
        __builtin_prefetch(W + (size_t)(n0 + (f8 >> 2)) * D_DIM + kpf + ((f8 & 3) << 3), 0, 3);
      }
    }

    v16bf ah[4], al[4];
#pragma unroll
    for (int mi = 0; mi < 4; ++mi) {
      ah[mi] = frag_a_lds(Xhi, wm * 64 + mi * 16, lane);
      al[mi] = frag_a_lds(Xlo, wm * 64 + mi * 16, lane);
    }
    v16bf bh[4], bl[4];
#pragma unroll
    for (int ni = 0; ni < 4; ++ni) {
      bh[ni] = frag_b_lds(Whi, wn * 64 + ni * 16, lane);
      bl[ni] = frag_b_lds(Wlo, wn * 64 + ni * 16, lane);
    }
#pragma unroll
    for (int mi = 0; mi < 4; ++mi)
#pragma unroll
      for (int ni = 0; ni < 4; ++ni) {
        acc[mi][ni] = wmma_bf16(ah[mi], bh[ni], acc[mi][ni]); // hi*hi
        acc[mi][ni] = wmma_bf16(al[mi], bh[ni], acc[mi][ni]); // lo*hi
        acc[mi][ni] = wmma_bf16(ah[mi], bl[ni], acc[mi][ni]); // hi*lo
      }
  }

  // ---- epilogue: out = acc + 2.0 * (T @ lora_b^T) + bias ----
  // T rows and lora_b rows are K-contiguous (16 bf16): load straight into
  // fragments with K zero-padded 16 -> 32.
  v16bf zero16 = {};
  v16bf tf[4];
#pragma unroll
  for (int mi = 0; mi < 4; ++mi) {
    const int row = m0 + wm * 64 + mi * 16 + (lane & 15);
    v8bf p = *(const v8bf*)(tmat + (size_t)row * R_DIM + 8 * half);
    v8bf z8 = {};
    tf[mi] = cat8(p, z8);               // A-frag: real K 0..15, K 16..31 = 0
  }
  v16bf bfr[4];
#pragma unroll
  for (int ni = 0; ni < 4; ++ni) {
    const int col = n0 + wn * 64 + ni * 16 + (lane & 15);
    const __bf16* bp = lora_b + (size_t)col * R_DIM;
    v8bf p0 = *(const v8bf*)(bp + 8 * half);
    v8bf p1 = {};
    v16bf full = cat8(p0, p1);
    bfr[ni] = full;                     // B-frag: real K 0..15, K 16..31 = 0
  }

#pragma unroll
  for (int mi = 0; mi < 4; ++mi)
#pragma unroll
    for (int ni = 0; ni < 4; ++ni) {
      v8f lacc = wmma_bf16(tf[mi], bfr[ni], zero);
      const int col = n0 + wn * 64 + ni * 16 + (lane & 15);
      const float bv = bias[col];
      v8f r = acc[mi][ni] + 2.0f * lacc + bv;     // LORA_ALPHA / R = 2.0
      const int rbase = m0 + wm * 64 + mi * 16 + 8 * half;
#pragma unroll
      for (int i = 0; i < 8; ++i)
        out[(size_t)(rbase + i) * D_DIM + col] = r[i];
    }
}

// ---------------------------------------------------------------------------
extern "C" void kernel_launch(void* const* d_in, const int* in_sizes, int n_in,
                              void* d_out, int out_size, void* d_ws, size_t ws_size,
                              hipStream_t stream) {
  (void)in_sizes; (void)n_in; (void)out_size; (void)ws_size;
  const float* x  = (const float*)d_in[0];
  const float* W  = (const float*)d_in[1];
  const float* b  = (const float*)d_in[2];
  const unsigned short* lora_a = (const unsigned short*)d_in[3]; // bf16 [16,4096]
  const unsigned short* lora_b = (const unsigned short*)d_in[4]; // bf16 [4096,16]
  float* out = (float*)d_out;
  unsigned short* tmat = (unsigned short*)d_ws;                  // bf16 [8192,16]

  lora_xa_kernel<<<T_DIM / 128, 256, 0, stream>>>(x, lora_a, tmat);
  lora_gemm_kernel<<<dim3(D_DIM / BN, T_DIM / BM), 256, 0, stream>>>(
      x, W, b, lora_b, tmat, out);
}
